// E_GCL_31928786878562
// MI455X (gfx1250) — matrix operand order repaired and checked by reading, not compiled, not run on hardware
//
#include <hip/hip_runtime.h>
#include <hip/hip_bf16.h>
#include <math.h>

// ---------------------------------------------------------------- constants
constexpr int N   = 20000;
constexpr int E   = 320000;
constexpr int H   = 128;
constexpr int HID = 128;
constexpr int C   = 3;
constexpr int B   = 100;

constexpr int K1E = 288;   // edge MLP layer1 K (258 padded to 32)
constexpr int K1V = 288;   // virtual-edge MLP layer1 K (260 padded)
constexpr int K1N = 672;   // node MLP layer1 K (641 padded)
constexpr int K1M = 256;   // virtual-node MLP layer1 K (exact)

typedef __bf16 bf16;
typedef __attribute__((ext_vector_type(16))) __bf16 v16bf;
typedef __attribute__((ext_vector_type(8)))  float  v8f;

// ------------------------------------------------- workspace layout (floats)
constexpr long OF_EAGG = 0;                               // [N,HID] edge_feat seg-sum
constexpr long OF_CNT  = OF_EAGG + (long)N * HID;         // [N]     edges per row
constexpr long OF_AGG  = OF_CNT  + N;                     // [N,3]   trans seg-sum
constexpr long OF_TRV  = OF_AGG  + (long)N * 3;           // [N,3]   trans_v accum
constexpr long OF_CSUM = OF_TRV  + (long)N * 3;           // [B,3]   coord seg-sum
constexpr long OF_CNTB = OF_CSUM + (long)B * 3;           // [B]     nodes per graph
constexpr long OF_MX   = OF_CNTB + B;                     // [B,3,3] mX
constexpr long OF_MAGG = OF_MX   + (long)B * 9;           // [B,HID,C] vef seg-sum
constexpr long OF_VAGG = OF_MAGG + (long)B * HID * C;     // [B,3,C] trans2 seg-sum
constexpr long F_TOT   = OF_VAGG + (long)B * 3 * C;
constexpr long F_TOTA  = ((F_TOT + 63) / 64) * 64;        // align bf16 region to 256B

// bf16 region (after floats): fragment-swizzled weights then vef
// swizzled weight size = Kpad*128 elements (same as dense), 8KB-aligned blocks
constexpr long WB_E1   = 0;                         // 288x128
constexpr long WB_E2   = WB_E1   + (long)K1E * HID; // 128x128
constexpr long WB_V1   = WB_E2   + (long)HID * HID; // 288x128
constexpr long WB_V2   = WB_V1   + (long)K1V * HID;
constexpr long WB_CR1  = WB_V2   + (long)HID * HID;
constexpr long WB_CRV1 = WB_CR1  + (long)HID * HID;
constexpr long WB_CVV1 = WB_CRV1 + (long)HID * HID;
constexpr long WB_N1   = WB_CVV1 + (long)HID * HID; // 672x128
constexpr long WB_N2   = WB_N1   + (long)K1N * HID;
constexpr long WB_M1   = WB_N2   + (long)HID * HID; // 256x128
constexpr long WB_M2   = WB_M1   + (long)K1M * HID;
constexpr long WB_VEF  = WB_M2   + (long)HID * HID; // [N,HID,C] bf16 vef
constexpr long WB_TOT  = WB_VEF  + (long)N * HID * C;

// output sections (float elements)
constexpr long OUT_NODE = 0;
constexpr long OUT_CRD  = OUT_NODE + (long)N * H;
constexpr long OUT_VN   = OUT_CRD  + (long)N * 3;
constexpr long OUT_VC   = OUT_VN   + (long)B * H * C;

__device__ __forceinline__ float silu_f(float x) { return x / (1.0f + __expf(-x)); }

// --------------------------------------------------------- WMMA fragment IO
// A 16x32 bf16 (MxK): lanes 0-15 row M, e0..15 -> K 0..15;
// lanes 16-31 same row, K 16..31.  One contiguous 32B run.
__device__ __forceinline__ v16bf load_A_frag(const bf16* sA, int lda, int kbase, int lid) {
    const int half = (lid >> 4) & 1;
    const int m    = lid & 15;
    v16bf a;
#pragma unroll
    for (int e = 0; e < 16; ++e) {
        int k = kbase + half * 16 + e;
        a[e] = sA[m * lda + k];
    }
    return a;
}
// B fragments stored pre-swizzled: fragment f, lane l -> 32 contiguous bytes.
__device__ __forceinline__ v16bf load_B_swz(const bf16* W, int frag, int lid) {
    const v16bf* p = (const v16bf*)W;
    return p[frag * 32 + lid];
}

template <int K>
__device__ __forceinline__ v8f gemm16(const bf16* sA, int lda, const bf16* W, int n0, int lid) {
    v8f acc = {};
    const int tb = n0 >> 4;
#pragma unroll
    for (int ks = 0; ks < K / 32; ++ks) {
        v16bf a = load_A_frag(sA, lda, ks * 32, lid);
        v16bf b = load_B_swz(W, ks * 8 + tb, lid);
        acc = __builtin_amdgcn_wmma_f32_16x16x32_bf16(false, a, false, b, (short)0, acc, false, false);
    }
    return acc;
}

// ------------------------------------------------------------- small kernels
__global__ void k_zero(float* p, long n) {
    long i = (long)blockIdx.x * blockDim.x + threadIdx.x;
    if (i < n) p[i] = 0.0f;
}

// convert f32 weight [K,128] into fragment-swizzled padded bf16
// dst[((ks*8 + t)*32 + l)*16 + e] = W[ks*32 + e + 16*(l>>4)][t*16 + (l&15)]
__global__ void k_cvt_swz(const float* src, bf16* dst, int K, int Kpad) {
    int i = blockIdx.x * blockDim.x + threadIdx.x;
    int tot = Kpad * HID;
    if (i >= tot) return;
    int e  = i & 15;
    int l  = (i >> 4) & 31;
    int t  = (i >> 9) & 7;
    int ks = i >> 12;
    int k  = ks * 32 + e + ((l >> 4) << 4);
    int n  = t * 16 + (l & 15);
    dst[i] = (k < K) ? (bf16)src[k * HID + n] : (bf16)0.0f;
}

__global__ void k_bstats(const float* coord, const int* batch, float* csum, float* cntB) {
    int n = blockIdx.x * blockDim.x + threadIdx.x;
    if (n >= N) return;
    int b = batch[n];
#pragma unroll
    for (int i = 0; i < 3; ++i) atomicAdd(&csum[b * 3 + i], coord[n * 3 + i]);
    atomicAdd(&cntB[b], 1.0f);
}

__global__ void k_mx(const float* vc, const float* csum, const float* cntB, float* mX) {
    int b = blockIdx.x * blockDim.x + threadIdx.x;
    if (b >= B) return;
    float inv = 1.0f / fmaxf(cntB[b], 1.0f);
    float m[3][3];
#pragma unroll
    for (int i = 0; i < 3; ++i) {
        float mean = csum[b * 3 + i] * inv;
#pragma unroll
        for (int c = 0; c < 3; ++c) m[i][c] = vc[(b * 3 + i) * C + c] - mean;
    }
#pragma unroll
    for (int c = 0; c < 3; ++c)
#pragma unroll
        for (int d = 0; d < 3; ++d) {
            float s = 0.0f;
#pragma unroll
            for (int i = 0; i < 3; ++i) s += m[i][c] * m[i][d];
            mX[b * 9 + c * 3 + d] = s;
        }
}

__global__ void k_coord_final(const float* coord, const float* aggs, const float* cnt,
                              const float* trv, float* out) {
    int n = blockIdx.x * blockDim.x + threadIdx.x;
    if (n >= N) return;
    float inv = 1.0f / fmaxf(cnt[n], 1.0f);
#pragma unroll
    for (int i = 0; i < 3; ++i)
        out[n * 3 + i] = coord[n * 3 + i] + aggs[n * 3 + i] * inv + trv[n * 3 + i];
}

__global__ void k_vcoord_final(const float* vc, const float* vagg, const float* cntB, float* out) {
    int i = blockIdx.x * blockDim.x + threadIdx.x;
    if (i >= B * 3 * C) return;
    int b = i / (3 * C);
    out[i] = vc[i] + vagg[i] / fmaxf(cntB[b], 1.0f);
}

// ------------------------------------------------- fused edge-MLP kernel
// 16 edges/block, 256 threads (8 waves -> 8 column tiles of 16).
__global__ __launch_bounds__(256) void k_edge(
    const float* __restrict__ nf, const float* __restrict__ coord,
    const float* __restrict__ eattr, const int* __restrict__ ei,
    const bf16* __restrict__ wb,
    const float* __restrict__ eb1, const float* __restrict__ eb2,
    const float* __restrict__ crb1, const float* __restrict__ crW2,
    float* eagg, float* cnt_row, float* agg_sum)
{
    __shared__ bf16  sA[16 * K1E];
    __shared__ bf16  sT1[16 * HID];
    __shared__ bf16  sT2[16 * HID];
    __shared__ float sScal[16];
    __shared__ float sCD[16 * 3];
    __shared__ float sRad[16];
    __shared__ float sEA[16];
    __shared__ int   sRow[16];
    __shared__ int   sCol[16];

    const int tid = threadIdx.x;
    const int lid = tid & 31;
    const int wv  = tid >> 5;
    const int n0  = wv * 16;
    const int half = lid >> 4;
    const int ncol = n0 + (lid & 15);

    if (tid < 16) {
        int e = blockIdx.x * 16 + tid;
        int r = ei[e], c = ei[E + e];
        sRow[tid] = r; sCol[tid] = c;
        float rad = 0.0f;
#pragma unroll
        for (int i = 0; i < 3; ++i) {
            float d = coord[r * 3 + i] - coord[c * 3 + i];
            sCD[tid * 3 + i] = d; rad += d * d;
        }
        sRad[tid] = rad;
        sEA[tid]  = eattr[e];
        sScal[tid] = 0.0f;
    }
    __syncthreads();

    for (int idx = tid; idx < 16 * K1E; idx += 256) {
        int m = idx / K1E, k = idx - m * K1E;
        float v;
        if      (k < 128) v = nf[sRow[m] * H + k];
        else if (k < 256) v = nf[sCol[m] * H + (k - 128)];
        else if (k == 256) v = sRad[m];
        else if (k == 257) v = sEA[m];
        else v = 0.0f;
        sA[idx] = (bf16)v;
    }
    __syncthreads();

    // layer 1: [16,288] @ eW1p -> silu
    {
        v8f acc = gemm16<K1E>(sA, K1E, wb + WB_E1, n0, lid);
        float bv = eb1[ncol];
#pragma unroll
        for (int r = 0; r < 8; ++r)
            sT1[(r + half * 8) * HID + ncol] = (bf16)silu_f(acc[r] + bv);
    }
    __syncthreads();

    // layer 2: edge_feat = silu(act1 @ eW2 + eb2); scatter into eagg
    {
        v8f acc = gemm16<HID>(sT1, HID, wb + WB_E2, n0, lid);
        float bv = eb2[ncol];
#pragma unroll
        for (int r = 0; r < 8; ++r) {
            int m = r + half * 8;
            float x = silu_f(acc[r] + bv);
            sT2[m * HID + ncol] = (bf16)x;
            atomicAdd(&eagg[(long)sRow[m] * HID + ncol], x);
        }
    }
    __syncthreads();

    // coord head: silu(edge_feat @ crW1 + crb1) . crW2 -> per-edge scalar
    {
        v8f acc = gemm16<HID>(sT2, HID, wb + WB_CR1, n0, lid);
        float bv = crb1[ncol];
        float w2 = crW2[ncol];
#pragma unroll
        for (int r = 0; r < 8; ++r)
            atomicAdd(&sScal[r + half * 8], silu_f(acc[r] + bv) * w2);
    }
    __syncthreads();

    if (tid < 16) {
        float s = sScal[tid];
        int row = sRow[tid];
#pragma unroll
        for (int i = 0; i < 3; ++i)
            atomicAdd(&agg_sum[row * 3 + i], sCD[tid * 3 + i] * s);
        atomicAdd(&cnt_row[row], 1.0f);
    }
}

// ------------------------------------------------- fused virtual-edge kernel
// rows r = n*C + c ; 16 rows/block
__global__ __launch_bounds__(256) void k_vedge(
    const float* __restrict__ nf, const float* __restrict__ coord,
    const float* __restrict__ vc, const float* __restrict__ vnf,
    const int* __restrict__ batch, const bf16* __restrict__ wb,
    const float* __restrict__ vb1, const float* __restrict__ vb2,
    const float* __restrict__ crvb1, const float* __restrict__ crvW2,
    const float* __restrict__ cvvb1, const float* __restrict__ cvvW2,
    const float* __restrict__ mX,
    bf16* vef_out, float* magg, float* transv, float* vagg)
{
    __shared__ bf16  sA[16 * K1V];
    __shared__ bf16  sT1[16 * HID];
    __shared__ bf16  sT2[16 * HID];
    __shared__ float sVCD[16 * 3];
    __shared__ float sRad[16];
    __shared__ float sSR[16];
    __shared__ float sSV[16];
    __shared__ int   sN[16], sB[16], sC[16];

    const int tid = threadIdx.x;
    const int lid = tid & 31;
    const int wv  = tid >> 5;
    const int n0  = wv * 16;
    const int half = lid >> 4;
    const int ncol = n0 + (lid & 15);

    if (tid < 16) {
        int r = blockIdx.x * 16 + tid;
        int n = r / C, c = r - n * C, b = batch[n];
        sN[tid] = n; sB[tid] = b; sC[tid] = c;
        float vr = 0.0f;
#pragma unroll
        for (int i = 0; i < 3; ++i) {
            float d = vc[(b * 3 + i) * C + c] - coord[n * 3 + i];
            sVCD[tid * 3 + i] = d; vr += d * d;
        }
        sRad[tid] = sqrtf(vr);
        sSR[tid] = 0.0f; sSV[tid] = 0.0f;
    }
    __syncthreads();

    for (int idx = tid; idx < 16 * K1V; idx += 256) {
        int m = idx / K1V, k = idx - m * K1V;
        int n = sN[m], b = sB[m], c = sC[m];
        float v;
        if      (k < 128) v = nf[n * H + k];
        else if (k < 256) v = vnf[(b * H + (k - 128)) * C + c];
        else if (k == 256) v = sRad[m];
        else if (k < 260) v = mX[b * 9 + c * 3 + (k - 257)];
        else v = 0.0f;
        sA[idx] = (bf16)v;
    }
    __syncthreads();

    // layer 1
    {
        v8f acc = gemm16<K1V>(sA, K1V, wb + WB_V1, n0, lid);
        float bv = vb1[ncol];
#pragma unroll
        for (int r = 0; r < 8; ++r)
            sT1[(r + half * 8) * HID + ncol] = (bf16)silu_f(acc[r] + bv);
    }
    __syncthreads();

    // layer 2: vef
    {
        v8f acc = gemm16<HID>(sT1, HID, wb + WB_V2, n0, lid);
        float bv = vb2[ncol];
#pragma unroll
        for (int r = 0; r < 8; ++r) {
            int m = r + half * 8;
            float x = silu_f(acc[r] + bv);
            sT2[m * HID + ncol] = (bf16)x;
            int n = sN[m], b = sB[m], c = sC[m];
            vef_out[((long)n * HID + ncol) * C + c] = (bf16)x;
            atomicAdd(&magg[((long)b * HID + ncol) * C + c], x);
        }
    }
    __syncthreads();

    // crv head -> scal_r
    {
        v8f acc = gemm16<HID>(sT2, HID, wb + WB_CRV1, n0, lid);
        float bv = crvb1[ncol], w2 = crvW2[ncol];
#pragma unroll
        for (int r = 0; r < 8; ++r)
            atomicAdd(&sSR[r + half * 8], silu_f(acc[r] + bv) * w2);
    }
    // cvv head -> scal_v
    {
        v8f acc = gemm16<HID>(sT2, HID, wb + WB_CVV1, n0, lid);
        float bv = cvvb1[ncol], w2 = cvvW2[ncol];
#pragma unroll
        for (int r = 0; r < 8; ++r)
            atomicAdd(&sSV[r + half * 8], silu_f(acc[r] + bv) * w2);
    }
    __syncthreads();

    if (tid < 16) {
        int n = sN[tid], b = sB[tid], c = sC[tid];
        float sr = sSR[tid], sv = sSV[tid];
#pragma unroll
        for (int i = 0; i < 3; ++i) {
            float d = sVCD[tid * 3 + i];
            atomicAdd(&transv[n * 3 + i], -(1.0f / C) * d * sr);
            atomicAdd(&vagg[(b * 3 + i) * C + c], d * sv);
        }
    }
}

// ------------------------------------------------- node-update MLP kernel
__global__ __launch_bounds__(256) void k_node(
    const float* __restrict__ nf, const float* __restrict__ nattr,
    const bf16* __restrict__ wb, const bf16* __restrict__ vef,
    const float* __restrict__ eagg, const float* __restrict__ cnt_row,
    const float* __restrict__ nb1, const float* __restrict__ nb2,
    float* out)
{
    __shared__ bf16  sA[16 * K1N];
    __shared__ bf16  sT1[16 * HID];
    __shared__ float sInv[16];

    const int tid = threadIdx.x;
    const int lid = tid & 31;
    const int wv  = tid >> 5;
    const int n0  = wv * 16;
    const int half = lid >> 4;
    const int ncol = n0 + (lid & 15);
    const int nbase = blockIdx.x * 16;

    if (tid < 16) sInv[tid] = 1.0f / fmaxf(cnt_row[nbase + tid], 1.0f);
    __syncthreads();

    for (int idx = tid; idx < 16 * K1N; idx += 256) {
        int m = idx / K1N, k = idx - m * K1N;
        int n = nbase + m;
        if      (k < 128) { sA[idx] = (bf16)nf[n * H + k]; }
        else if (k < 256) { sA[idx] = (bf16)(eagg[(long)n * HID + (k - 128)] * sInv[m]); }
        else if (k < 640) { sA[idx] = vef[(long)n * HID * C + (k - 256)]; }
        else if (k == 640){ sA[idx] = (bf16)nattr[n]; }
        else              { sA[idx] = (bf16)0.0f; }
    }
    __syncthreads();

    {
        v8f acc = gemm16<K1N>(sA, K1N, wb + WB_N1, n0, lid);
        float bv = nb1[ncol];
#pragma unroll
        for (int r = 0; r < 8; ++r)
            sT1[(r + half * 8) * HID + ncol] = (bf16)silu_f(acc[r] + bv);
    }
    __syncthreads();

    {
        v8f acc = gemm16<HID>(sT1, HID, wb + WB_N2, n0, lid);
        float bv = nb2[ncol];
#pragma unroll
        for (int r = 0; r < 8; ++r) {
            int n = nbase + r + half * 8;
            out[(long)n * H + ncol] = nf[(long)n * H + ncol] + acc[r] + bv;
        }
    }
}

// ------------------------------------------------- virtual-node MLP kernel
// rows rb = b*C + c, total B*C = 300
__global__ __launch_bounds__(256) void k_vnode(
    const float* __restrict__ vnf, const bf16* __restrict__ wb,
    const float* __restrict__ magg, const float* __restrict__ cntB,
    const float* __restrict__ mb1, const float* __restrict__ mb2,
    float* out)
{
    __shared__ bf16  sA[16 * K1M];
    __shared__ bf16  sT1[16 * HID];
    __shared__ float sInv[16];
    __shared__ int   sB[16], sC[16], sOK[16];

    const int tid = threadIdx.x;
    const int lid = tid & 31;
    const int wv  = tid >> 5;
    const int n0  = wv * 16;
    const int half = lid >> 4;
    const int ncol = n0 + (lid & 15);
    const int rbase = blockIdx.x * 16;

    if (tid < 16) {
        int rb = rbase + tid;
        if (rb < B * C) {
            int b = rb / C;
            sB[tid] = b; sC[tid] = rb - b * C; sOK[tid] = 1;
            sInv[tid] = 1.0f / fmaxf(cntB[b], 1.0f);
        } else { sB[tid] = 0; sC[tid] = 0; sOK[tid] = 0; sInv[tid] = 0.0f; }
    }
    __syncthreads();

    for (int idx = tid; idx < 16 * K1M; idx += 256) {
        int m = idx / K1M, k = idx - m * K1M;
        float v = 0.0f;
        if (sOK[m]) {
            int b = sB[m], c = sC[m];
            if (k < 128) v = vnf[(b * H + k) * C + c];
            else         v = magg[((long)b * HID + (k - 128)) * C + c] * sInv[m];
        }
        sA[idx] = (bf16)v;
    }
    __syncthreads();

    {
        v8f acc = gemm16<K1M>(sA, K1M, wb + WB_M1, n0, lid);
        float bv = mb1[ncol];
#pragma unroll
        for (int r = 0; r < 8; ++r)
            sT1[(r + half * 8) * HID + ncol] = (bf16)silu_f(acc[r] + bv);
    }
    __syncthreads();

    {
        v8f acc = gemm16<HID>(sT1, HID, wb + WB_M2, n0, lid);
        float bv = mb2[ncol];
#pragma unroll
        for (int r = 0; r < 8; ++r) {
            int m = r + half * 8;
            int rb = rbase + m;
            if (rb < B * C) {
                int b = rb / C, c = rb - (rb / C) * C;
                long o = ((long)b * H + ncol) * C + c;
                out[o] = vnf[o] + acc[r] + bv;
            }
        }
    }
}

// ---------------------------------------------------------------- launcher
extern "C" void kernel_launch(void* const* d_in, const int* in_sizes, int n_in,
                              void* d_out, int out_size, void* d_ws, size_t ws_size,
                              hipStream_t stream) {
    const float* nf     = (const float*)d_in[0];
    const float* coord  = (const float*)d_in[1];
    const float* vc     = (const float*)d_in[2];
    const float* vnf    = (const float*)d_in[3];
    const float* eattr  = (const float*)d_in[4];
    const float* nattr  = (const float*)d_in[5];
    const int*   ei     = (const int*)d_in[6];
    const int*   batch  = (const int*)d_in[7];
    // params in dict order
    const float* eW1  = (const float*)d_in[8];
    const float* eb1  = (const float*)d_in[9];
    const float* eW2  = (const float*)d_in[10];
    const float* eb2  = (const float*)d_in[11];
    const float* vW1  = (const float*)d_in[12];
    const float* vb1  = (const float*)d_in[13];
    const float* vW2  = (const float*)d_in[14];
    const float* vb2  = (const float*)d_in[15];
    const float* crW1 = (const float*)d_in[16];
    const float* crb1 = (const float*)d_in[17];
    const float* crW2 = (const float*)d_in[18];
    const float* crvW1= (const float*)d_in[19];
    const float* crvb1= (const float*)d_in[20];
    const float* crvW2= (const float*)d_in[21];
    const float* cvvW1= (const float*)d_in[22];
    const float* cvvb1= (const float*)d_in[23];
    const float* cvvW2= (const float*)d_in[24];
    const float* nW1  = (const float*)d_in[25];
    const float* nb1  = (const float*)d_in[26];
    const float* nW2  = (const float*)d_in[27];
    const float* nb2  = (const float*)d_in[28];
    const float* mW1  = (const float*)d_in[29];
    const float* mb1  = (const float*)d_in[30];
    const float* mW2  = (const float*)d_in[31];
    const float* mb2  = (const float*)d_in[32];

    float* fws  = (float*)d_ws;
    bf16*  wb   = (bf16*)(fws + F_TOTA);
    bf16*  vefb = wb + WB_VEF;
    float* out  = (float*)d_out;

    float* eagg   = fws + OF_EAGG;
    float* cnt    = fws + OF_CNT;
    float* aggs   = fws + OF_AGG;
    float* trv    = fws + OF_TRV;
    float* csum   = fws + OF_CSUM;
    float* cntB   = fws + OF_CNTB;
    float* mX     = fws + OF_MX;
    float* magg   = fws + OF_MAGG;
    float* vagg   = fws + OF_VAGG;

    // 1) zero accumulators
    {
        long n = F_TOT;
        k_zero<<<(unsigned)((n + 255) / 256), 256, 0, stream>>>(fws, n);
    }
    // 2) convert + pad weights into fragment-swizzled bf16
    auto cvt = [&](const float* src, long off, int K, int Kpad) {
        int tot = Kpad * HID;
        k_cvt_swz<<<(tot + 255) / 256, 256, 0, stream>>>(src, wb + off, K, Kpad);
    };
    cvt(eW1, WB_E1, 258, K1E);
    cvt(eW2, WB_E2, 128, 128);
    cvt(vW1, WB_V1, 260, K1V);
    cvt(vW2, WB_V2, 128, 128);
    cvt(crW1, WB_CR1, 128, 128);
    cvt(crvW1, WB_CRV1, 128, 128);
    cvt(cvvW1, WB_CVV1, 128, 128);
    cvt(nW1, WB_N1, 641, K1N);
    cvt(nW2, WB_N2, 128, 128);
    cvt(mW1, WB_M1, 256, K1M);
    cvt(mW2, WB_M2, 128, 128);
    // 3) batch stats + mX
    k_bstats<<<(N + 255) / 256, 256, 0, stream>>>(coord, batch, csum, cntB);
    k_mx<<<(B + 63) / 64, 64, 0, stream>>>(vc, csum, cntB, mX);
    // 4) fused edge MLP + head (3 WMMA GEMMs per tile)
    k_edge<<<E / 16, 256, 0, stream>>>(nf, coord, eattr, ei, wb, eb1, eb2,
                                       crb1, crW2, eagg, cnt, aggs);
    // 5) fused virtual-edge MLP + 2 heads (4 WMMA GEMMs per tile)
    k_vedge<<<(N * C) / 16, 256, 0, stream>>>(nf, coord, vc, vnf, batch, wb,
                                              vb1, vb2, crvb1, crvW2, cvvb1, cvvW2,
                                              mX, vefb, magg, trv, vagg);
    // 6) node update MLP -> d_out[OUT_NODE]
    k_node<<<N / 16, 256, 0, stream>>>(nf, nattr, wb, vefb, eagg, cnt,
                                       nb1, nb2, out + OUT_NODE);
    // 7) virtual node MLP -> d_out[OUT_VN]
    k_vnode<<<(B * C + 15) / 16, 256, 0, stream>>>(vnf, wb, magg, cntB,
                                                   mb1, mb2, out + OUT_VN);
    // 8) coordinate finishers
    k_coord_final<<<(N + 255) / 256, 256, 0, stream>>>(coord, aggs, cnt, trv, out + OUT_CRD);
    k_vcoord_final<<<(B * 3 * C + 255) / 256, 256, 0, stream>>>(vc, vagg, cntB, out + OUT_VC);
}
